// DeformNet_833223656365
// MI455X (gfx1250) — matrix operand order repaired and checked by reading, not compiled, not run on hardware
//
#include <hip/hip_runtime.h>
#include <hip/hip_bf16.h>

// ---------------------------------------------------------------------------
// DeformNet forward for MI455X (gfx1250), wave32 + WMMA f16 implicit GEMM.
// Dense convs (conv2/3/4, ~500 GFLOP total) run as implicit GEMM on
// v_wmma_f32_16x16x32_f16 with full M register blocking: each wave owns 16
// spatial points x ALL output channels, so one activation (B) fragment feeds
// MT = O/16 WMMAs into independent accumulators.
// ---------------------------------------------------------------------------

typedef _Float16 h16;
typedef __attribute__((ext_vector_type(8)))  _Float16 v8h;
typedef __attribute__((ext_vector_type(16))) _Float16 v16h;
typedef __attribute__((ext_vector_type(8)))  float    v8f;

#define DD    28
#define SS    (28*28*28)          // 21952 spatial points
#define DP    30                  // padded dim
#define SPAD  (30*30*30)          // 27000
#define BATCH 16
#define ROWS  (BATCH*SS)          // 351232

// -------------------------------- utility ----------------------------------

__global__ void k_zero(unsigned int* p, size_t n) {
  size_t i = (size_t)blockIdx.x * blockDim.x + threadIdx.x;
  if (i < n) p[i] = 0u;
}

// copy x (b,1,28,28,28) into zero-padded xp (b,30,30,30)
__global__ void k_padfill(const float* __restrict__ x, float* __restrict__ xp) {
  int i = blockIdx.x * blockDim.x + threadIdx.x;
  if (i >= BATCH * SS) return;
  int b = i / SS, s = i % SS;
  int d = s / 784, h = (s / 28) % 28, w = s % 28;
  xp[(((size_t)b * DP + d + 1) * DP + h + 1) * DP + (w + 1)] = x[i];
}

// ----------------------------- offsets conv --------------------------------
// offsets = conv3d(x, off_w, off_b), out (b,81,28,28,28) written to d_out+160
__global__ void k_offsets(const float* __restrict__ xp, const float* __restrict__ off_w,
                          const float* __restrict__ off_b, float* __restrict__ out) {
  __shared__ float sw[81 * 27];
  __shared__ float sb[81];
  for (int i = threadIdx.x; i < 81 * 27; i += blockDim.x) sw[i] = off_w[i];
  for (int i = threadIdx.x; i < 81;      i += blockDim.x) sb[i] = off_b[i];
  __syncthreads();
  int i = blockIdx.x * blockDim.x + threadIdx.x;
  if (i >= BATCH * SS) return;
  int b = i / SS, s = i % SS;
  int d = s / 784, h = (s / 28) % 28, w = s % 28;
  float xv[27];
  const float* base = xp + (((size_t)b * DP + d) * DP + h) * DP + w;
#pragma unroll
  for (int kd = 0; kd < 3; kd++)
#pragma unroll
    for (int kh = 0; kh < 3; kh++)
#pragma unroll
      for (int kw = 0; kw < 3; kw++)
        xv[(kd * 3 + kh) * 3 + kw] = base[(kd * DP + kh) * DP + kw];
  for (int o = 0; o < 81; o++) {
    float a = sb[o];
#pragma unroll
    for (int t = 0; t < 27; t++) a = fmaf(sw[o * 27 + t], xv[t], a);
    out[((size_t)(b * 81 + o)) * SS + s] = a;
  }
}

// ----------------------------- deformable conv -----------------------------
// relu(deform_conv3d(x, offsets, dconv_w)) -> act [ROWS][32] channels-last
__global__ void k_deform(const float* __restrict__ xp, const float* __restrict__ offs,
                         const float* __restrict__ dw, float* __restrict__ act) {
  __shared__ float sw[32 * 27];
  for (int i = threadIdx.x; i < 32 * 27; i += blockDim.x) sw[i] = dw[i];
  __syncthreads();
  int i = blockIdx.x * blockDim.x + threadIdx.x;
  if (i >= BATCH * SS) return;
  int b = i / SS, s = i % SS;
  int d = s / 784, h = (s / 28) % 28, w = s % 28;
  const float* xb = xp + (size_t)b * SPAD;
  float accn[27];
#pragma unroll
  for (int n = 0; n < 27; n++) {
    float od = offs[((size_t)(b * 81 + n)) * SS + s];
    float oh = offs[((size_t)(b * 81 + 27 + n)) * SS + s];
    float ow = offs[((size_t)(b * 81 + 54 + n)) * SS + s];
    // base (d+1) + kernel offset (n/9 - 1) + learned offset
    float pd = (float)(d + (n / 9))        + od;
    float ph = (float)(h + ((n / 3) % 3))  + oh;
    float pw = (float)(w + (n % 3))        + ow;

    float q0d = floorf(pd), q0h = floorf(ph), q0w = floorf(pw);
    float q0cd = fminf(fmaxf(q0d, 0.f), 29.f);
    float q0ch = fminf(fmaxf(q0h, 0.f), 29.f);
    float q0cw = fminf(fmaxf(q0w, 0.f), 29.f);
    float q1cd = fminf(fmaxf(q0d + 1.f, 0.f), 29.f);
    float q1ch = fminf(fmaxf(q0h + 1.f, 0.f), 29.f);
    float q1cw = fminf(fmaxf(q0w + 1.f, 0.f), 29.f);
    float pcd = fminf(fmaxf(pd, 0.f), 29.f);
    float pch = fminf(fmaxf(ph, 0.f), 29.f);
    float pcw = fminf(fmaxf(pw, 0.f), 29.f);
    float wlod = 1.f + q0cd - pcd, whid = 1.f - q1cd + pcd;
    float wloh = 1.f + q0ch - pch, whih = 1.f - q1ch + pch;
    float wlow = 1.f + q0cw - pcw, whiw = 1.f - q1cw + pcw;
    int i0d = (int)q0cd, i1d = (int)q1cd;
    int i0h = (int)q0ch, i1h = (int)q1ch;
    int i0w = (int)q0cw, i1w = (int)q1cw;

    float a = 0.f;
    a = fmaf(wlod * wloh * wlow, xb[(i0d * DP + i0h) * DP + i0w], a);
    a = fmaf(wlod * wloh * whiw, xb[(i0d * DP + i0h) * DP + i1w], a);
    a = fmaf(wlod * whih * wlow, xb[(i0d * DP + i1h) * DP + i0w], a);
    a = fmaf(wlod * whih * whiw, xb[(i0d * DP + i1h) * DP + i1w], a);
    a = fmaf(whid * wloh * wlow, xb[(i1d * DP + i0h) * DP + i0w], a);
    a = fmaf(whid * wloh * whiw, xb[(i1d * DP + i0h) * DP + i1w], a);
    a = fmaf(whid * whih * wlow, xb[(i1d * DP + i1h) * DP + i0w], a);
    a = fmaf(whid * whih * whiw, xb[(i1d * DP + i1h) * DP + i1w], a);
    accn[n] = a;
  }
  for (int o = 0; o < 32; o++) {
    float v = 0.f;
#pragma unroll
    for (int t = 0; t < 27; t++) v = fmaf(sw[o * 27 + t], accn[t], v);
    act[(size_t)i * 32 + o] = fmaxf(v, 0.f);
  }
}

// ------------------------- weight transpose to f16 -------------------------
// w f32 [O][C][27] (OIDHW) -> f16 [O][tap*C + c]
__global__ void k_wcvt(const float* __restrict__ w, h16* __restrict__ out, int O, int C) {
  int i = blockIdx.x * blockDim.x + threadIdx.x;
  int total = O * C * 27;
  if (i >= total) return;
  int t = i % 27;
  int c = (i / 27) % C;
  int o = i / (27 * C);
  out[(size_t)o * (27 * C) + t * C + c] = (h16)w[i];
}

// ------------------------- implicit-GEMM WMMA conv -------------------------
// in:  padded f16 activations [B][30][30][30][C]   (C multiple of 32)
// wk:  f16 weights [O][27*C], K index = tap*C + c  (O multiple of 16)
// out: relu(conv + bias) f32, channels-last [ROWS][O]
// One wave: 16 spatial points x O output channels. Each B (activation)
// fragment is loaded once and feeds MT = O/16 WMMAs (independent accums).
template <int C, int O>
__global__ void __launch_bounds__(256)
k_conv_wmma(const h16* __restrict__ in, const h16* __restrict__ wk,
            const float* __restrict__ bias, float* __restrict__ out) {
  constexpr int MT  = O / 16;   // M tiles, all owned by this wave
  constexpr int CPT = C / 32;   // K chunks per tap
  const int lane  = threadIdx.x & 31;
  const int gw    = blockIdx.x * (blockDim.x >> 5) + (threadIdx.x >> 5);
  const int sTile = gw % (SS / 16);
  const int b     = gw / (SS / 16);

  const int col   = lane & 15;   // N column / A row (per ISA fragment layout)
  const int khalf = lane >> 4;

  const int s = sTile * 16 + col;
  const int d = s / 784, h = (s / 28) % 28, w = s % 28;
  const size_t colBase = ((((size_t)b * DP + d) * DP + h) * DP + w) * C;
  // A row base for this lane: row = m*16 + col; elems 0..15 <-> K=khalf*16..+15
  // contiguous (matches B fragment order)
  const h16* arow = wk + (size_t)col * (27 * C) + khalf * 16;

  v8f acc[MT] = {};

#pragma unroll 1
  for (int tap = 0; tap < 27; ++tap) {
    const int kd = tap / 9, kh = (tap / 3) % 3, kw = tap % 3;
    const h16* bbase = in + colBase + (size_t)((kd * DP + kh) * DP + kw) * C
                          + khalf * 16;
#pragma unroll
    for (int part = 0; part < CPT; ++part) {
      const int kc = tap * CPT + part;
      // B fragment: lane col = n; K = khalf*16 .. +15 contiguous channels
      const h16* bp = bbase + part * 32;
      v8h blo = *(const v8h*)bp;
      v8h bhi = *(const v8h*)(bp + 8);
      v16h Bm = __builtin_shufflevector(blo, bhi, 0, 1, 2, 3, 4, 5, 6, 7,
                                        8, 9, 10, 11, 12, 13, 14, 15);
#pragma unroll
      for (int m = 0; m < MT; m++) {
        const h16* ap = arow + (size_t)m * 16 * (27 * C) + kc * 32;
        v8h alo = *(const v8h*)ap;
        v8h ahi = *(const v8h*)(ap + 8);
        v16h A = __builtin_shufflevector(alo, ahi, 0, 1, 2, 3, 4, 5, 6, 7,
                                         8, 9, 10, 11, 12, 13, 14, 15);
        acc[m] = __builtin_amdgcn_wmma_f32_16x16x32_f16(false, A, false, Bm,
                                                        (short)0, acc[m],
                                                        false, false);
      }
    }
  }

  // C/D layout: lanes 0-15 rows 0..7, lanes 16-31 rows 8..15; N = lane&15
  const int rowoff = khalf * 8;
  float* op0 = out + ((size_t)b * SS + s) * O;
#pragma unroll
  for (int m = 0; m < MT; m++) {
    const int rb = m * 16 + rowoff;
    float4 r0, r1;
    r0.x = fmaxf(acc[m][0] + bias[rb + 0], 0.f);
    r0.y = fmaxf(acc[m][1] + bias[rb + 1], 0.f);
    r0.z = fmaxf(acc[m][2] + bias[rb + 2], 0.f);
    r0.w = fmaxf(acc[m][3] + bias[rb + 3], 0.f);
    r1.x = fmaxf(acc[m][4] + bias[rb + 4], 0.f);
    r1.y = fmaxf(acc[m][5] + bias[rb + 5], 0.f);
    r1.z = fmaxf(acc[m][6] + bias[rb + 6], 0.f);
    r1.w = fmaxf(acc[m][7] + bias[rb + 7], 0.f);
    *(float4*)(op0 + rb) = r0;
    *(float4*)(op0 + rb + 4) = r1;
  }
}

// --------------------------- BN statistics & apply -------------------------

template <int C>
__global__ void k_chanreduce(const float* __restrict__ act,
                             float* __restrict__ sum, float* __restrict__ sumsq) {
  int c = threadIdx.x;   // blockDim.x == C
  float s = 0.f, q = 0.f;
  for (long r = blockIdx.x; r < ROWS; r += gridDim.x) {
    float x = act[(size_t)r * C + c];
    s += x;
    q = fmaf(x, x, q);
  }
  atomicAdd(&sum[c], s);
  atomicAdd(&sumsq[c], q);
}

__global__ void k_bnfinal(const float* sum, const float* sumsq, const float* g,
                          const float* bb, float* scale, float* shift, int C) {
  int c = threadIdx.x;
  if (c >= C) return;
  float inv = 1.f / (float)ROWS;
  float m = sum[c] * inv;
  float v = fmaf(-m, m, sumsq[c] * inv);       // biased variance
  float sc = g[c] * rsqrtf(v + 1e-5f);
  scale[c] = sc;
  shift[c] = fmaf(-m, sc, bb[c]);
}

// y = x*scale+shift, f16-convert into padded channels-last buffer
template <int C>
__global__ void k_bnapply(const float* __restrict__ act, const float* __restrict__ scale,
                          const float* __restrict__ shift, h16* __restrict__ P) {
  size_t i = (size_t)blockIdx.x * blockDim.x + threadIdx.x;
  size_t total = (size_t)ROWS * C;
  if (i >= total) return;
  int c = (int)(i & (C - 1));            // C is a power of two
  size_t rs = i / C;
  int s = (int)(rs % SS);
  int b = (int)(rs / SS);
  int d = s / 784, h = (s / 28) % 28, w = s % 28;
  float y = fmaf(act[i], scale[c], shift[c]);
  P[((((size_t)b * DP + d + 1) * DP + h + 1) * DP + (w + 1)) * C + c] = (h16)y;
}

// ------------------------------ head -----------------------------------

// per-(b,c) spatial sums of act4 [ROWS][128] -> sumbc[16][128]
__global__ void k_bcreduce(const float* __restrict__ act, float* __restrict__ sumbc) {
  int b = blockIdx.x >> 5;
  int chunk = blockIdx.x & 31;
  int c = threadIdx.x;          // 128
  const int CH = SS / 32;       // 686
  float s = 0.f;
  for (int k = 0; k < CH; k++) {
    int ss = chunk * CH + k;
    s += act[(((size_t)b * SS) + ss) * 128 + c];
  }
  atomicAdd(&sumbc[b * 128 + c], s);
}

// BN4 folded into pooled mean, FC, log_softmax -> d_out[0..159]
__global__ void k_final(const float* sum, const float* sumsq, const float* g,
                        const float* bb, const float* sumbc, const float* fcw,
                        const float* fcb, float* __restrict__ out) {
  __shared__ float pooled[16][128];
  __shared__ float logits[16][10];
  int t = threadIdx.x;
  if (t < 128) {
    float inv = 1.f / (float)ROWS;
    float m = sum[t] * inv;
    float v = fmaf(-m, m, sumsq[t] * inv);
    float sc = g[t] * rsqrtf(v + 1e-5f);
    float sh = fmaf(-m, sc, bb[t]);
    for (int b = 0; b < 16; b++)
      pooled[b][t] = fmaf(sumbc[b * 128 + t] * (1.f / (float)SS), sc, sh);
  }
  __syncthreads();
  if (t < 160) {
    int b = t / 10, o = t % 10;
    float l = fcb[o];
    for (int c = 0; c < 128; c++) l = fmaf(pooled[b][c], fcw[o * 128 + c], l);
    logits[b][o] = l;
  }
  __syncthreads();
  if (t < 16) {
    float mx = -1e30f;
    for (int o = 0; o < 10; o++) mx = fmaxf(mx, logits[t][o]);
    float se = 0.f;
    for (int o = 0; o < 10; o++) se += expf(logits[t][o] - mx);
    float lse = mx + logf(se);
    for (int o = 0; o < 10; o++) out[t * 10 + o] = logits[t][o] - lse;
  }
}

// ------------------------------- launcher ----------------------------------

extern "C" void kernel_launch(void* const* d_in, const int* in_sizes, int n_in,
                              void* d_out, int out_size, void* d_ws, size_t ws_size,
                              hipStream_t stream) {
  (void)in_sizes; (void)n_in; (void)out_size; (void)ws_size;
  const float* x       = (const float*)d_in[0];
  const float* off_w   = (const float*)d_in[1];
  const float* off_b   = (const float*)d_in[2];
  const float* dconv_w = (const float*)d_in[3];
  const float* bn1_g   = (const float*)d_in[4];
  const float* bn1_b   = (const float*)d_in[5];
  const float* conv2_w = (const float*)d_in[6];
  const float* conv2_b = (const float*)d_in[7];
  const float* bn2_g   = (const float*)d_in[8];
  const float* bn2_b   = (const float*)d_in[9];
  const float* conv3_w = (const float*)d_in[10];
  const float* conv3_b = (const float*)d_in[11];
  const float* bn3_g   = (const float*)d_in[12];
  const float* bn3_b   = (const float*)d_in[13];
  const float* conv4_w = (const float*)d_in[14];
  const float* conv4_b = (const float*)d_in[15];
  const float* bn4_g   = (const float*)d_in[16];
  const float* bn4_b   = (const float*)d_in[17];
  const float* fc_w    = (const float*)d_in[18];
  const float* fc_b    = (const float*)d_in[19];

  float* out = (float*)d_out;
  float* offsets = out + 160;   // second return value, flat after logits

  // ---- workspace carving (total ~350 MB) ----
  size_t woff = 0;
  auto carve = [&](size_t bytes) -> char* {
    char* p = (char*)d_ws + woff;
    woff += (bytes + 255) & ~(size_t)255;
    return p;
  };
  float* xp  = (float*)carve((size_t)BATCH * SPAD * 4);           // 1.7 MB
  float* act = (float*)carve((size_t)ROWS * 128 * 4);             // 180 MB
  h16*  Pa   = (h16*)carve((size_t)BATCH * SPAD * 128 * 2);       // 110 MB (C=32 then C=128)
  h16*  Pb   = (h16*)carve((size_t)BATCH * SPAD * 64 * 2);        // 55 MB  (C=64)
  h16*  Wb2  = (h16*)carve((size_t)64 * 864 * 2);
  h16*  Wb3  = (h16*)carve((size_t)128 * 1728 * 2);
  h16*  Wb4  = (h16*)carve((size_t)128 * 3456 * 2);
  float* chS = (float*)carve(128 * 4);
  float* chQ = (float*)carve(128 * 4);
  float* sc  = (float*)carve(128 * 4);
  float* sh  = (float*)carve(128 * 4);
  float* sbc = (float*)carve(16 * 128 * 4);

  auto NB = [](size_t n) { return (unsigned)((n + 255) / 256); };
  const int PTS = BATCH * SS;
  const int CONV_BLOCKS = (BATCH * (SS / 16)) / 8;   // 2744: 8 waves/block

  // stage 0: padded x, offsets conv, deformable conv (act = relu, [ROWS][32])
  k_zero<<<NB((size_t)BATCH * SPAD), 256, 0, stream>>>((unsigned*)xp, (size_t)BATCH * SPAD);
  k_padfill<<<NB(PTS), 256, 0, stream>>>(x, xp);
  k_offsets<<<NB(PTS), 256, 0, stream>>>(xp, off_w, off_b, offsets);
  k_deform<<<NB(PTS), 256, 0, stream>>>(xp, offsets, dconv_w, act);

  // weight transposes (independent)
  k_wcvt<<<NB(64 * 32 * 27), 256, 0, stream>>>(conv2_w, Wb2, 64, 32);
  k_wcvt<<<NB(128 * 64 * 27), 256, 0, stream>>>(conv3_w, Wb3, 128, 64);
  k_wcvt<<<NB(128 * 128 * 27), 256, 0, stream>>>(conv4_w, Wb4, 128, 128);

  // BN1 -> Pa (C=32)
  k_zero<<<1, 256, 0, stream>>>((unsigned*)chS, 128);
  k_zero<<<1, 256, 0, stream>>>((unsigned*)chQ, 128);
  k_chanreduce<32><<<512, 32, 0, stream>>>(act, chS, chQ);
  k_bnfinal<<<1, 32, 0, stream>>>(chS, chQ, bn1_g, bn1_b, sc, sh, 32);
  k_zero<<<NB((size_t)BATCH * SPAD * 32 / 2), 256, 0, stream>>>((unsigned*)Pa, (size_t)BATCH * SPAD * 32 / 2);
  k_bnapply<32><<<NB((size_t)ROWS * 32), 256, 0, stream>>>(act, sc, sh, Pa);

  // conv2: 32 -> 64
  k_conv_wmma<32, 64><<<CONV_BLOCKS, 256, 0, stream>>>(Pa, Wb2, conv2_b, act);

  // BN2 -> Pb (C=64)
  k_zero<<<1, 256, 0, stream>>>((unsigned*)chS, 128);
  k_zero<<<1, 256, 0, stream>>>((unsigned*)chQ, 128);
  k_chanreduce<64><<<512, 64, 0, stream>>>(act, chS, chQ);
  k_bnfinal<<<1, 64, 0, stream>>>(chS, chQ, bn2_g, bn2_b, sc, sh, 64);
  k_zero<<<NB((size_t)BATCH * SPAD * 64 / 2), 256, 0, stream>>>((unsigned*)Pb, (size_t)BATCH * SPAD * 64 / 2);
  k_bnapply<64><<<NB((size_t)ROWS * 64), 256, 0, stream>>>(act, sc, sh, Pb);

  // conv3: 64 -> 128
  k_conv_wmma<64, 128><<<CONV_BLOCKS, 256, 0, stream>>>(Pb, Wb3, conv3_b, act);

  // BN3 -> Pa (C=128, layer-1 data already consumed)
  k_zero<<<1, 256, 0, stream>>>((unsigned*)chS, 128);
  k_zero<<<1, 256, 0, stream>>>((unsigned*)chQ, 128);
  k_chanreduce<128><<<512, 128, 0, stream>>>(act, chS, chQ);
  k_bnfinal<<<1, 128, 0, stream>>>(chS, chQ, bn3_g, bn3_b, sc, sh, 128);
  k_zero<<<NB((size_t)BATCH * SPAD * 128 / 2), 256, 0, stream>>>((unsigned*)Pa, (size_t)BATCH * SPAD * 128 / 2);
  k_bnapply<128><<<NB((size_t)ROWS * 128), 256, 0, stream>>>(act, sc, sh, Pa);

  // conv4: 128 -> 128
  k_conv_wmma<128, 128><<<CONV_BLOCKS, 256, 0, stream>>>(Pa, Wb4, conv4_b, act);

  // BN4 stats + per-batch pooled sums + head
  k_zero<<<1, 256, 0, stream>>>((unsigned*)chS, 128);
  k_zero<<<1, 256, 0, stream>>>((unsigned*)chQ, 128);
  k_zero<<<NB(16 * 128), 256, 0, stream>>>((unsigned*)sbc, 16 * 128);
  k_chanreduce<128><<<512, 128, 0, stream>>>(act, chS, chQ);
  k_bcreduce<<<512, 128, 0, stream>>>(act, sbc);
  k_final<<<1, 256, 0, stream>>>(chS, chQ, bn4_g, bn4_b, sbc, fc_w, fc_b, out);
}
